// SequentialRBN_35253091565583
// MI455X (gfx1250) — matrix operand order repaired and checked by reading, not compiled, not run on hardware
//
#include <hip/hip_runtime.h>

typedef __attribute__((ext_vector_type(16))) __bf16 bf16x16;
typedef __attribute__((ext_vector_type(8)))  float  f32x8;

#define NSEQ 64
#define BATCH 8
#define KDIM 64
#define TDIM 128

#define LT_S 72     // Lt row stride (bf16), 16B aligned, bank-spread
#define RT_S 72     // Rt row stride
#define PR_S 2056   // pairLDS row stride: 16 x (xy half 2048) + pad (4112B)

__device__ __forceinline__ int cellOff(int i, int j, int b) {
    // chart[i][j][b][k], k contiguous (64 elems)
    return (((i * (NSEQ + 1)) + j) * BATCH + b) * KDIM;
}

union Frag {
    bf16x16 v;
    uint4   u[2];
};
union Pack {
    uint4  q;
    __bf16 h[8];
};

// ------------------------------------------------- rule -> bf16, permuted
// rule_tf[a][ h*2048 + y*32 + (x&31) ] = rule[a][x][y],  h = x>>5
// Matches pairLDS's inner ordering so phase B contracts correctly.
__global__ void convert_rule_kernel(const float* __restrict__ rule,
                                    __bf16* __restrict__ rule_tf) {
    int i = blockIdx.x * blockDim.x + threadIdx.x;   // 0 .. 262143
    int a = i >> 12;
    int x = (i >> 6) & 63;
    int y = i & 63;
    int o = a * 4096 + (x >> 5) * 2048 + y * 32 + (x & 31);
    rule_tf[o] = (__bf16)rule[i];
}

// ------------------------------------------------- span length 1
__global__ void span1_kernel(const float* __restrict__ seq,
                             const float* __restrict__ emit,
                             __bf16* __restrict__ chart) {
    int n = blockIdx.x;
    int b = blockIdx.y;
    int a = threadIdx.x;
    const float* sp = seq + (b * NSEQ + n) * TDIM;
    const float* ep = emit + a * TDIM;
    float acc = 0.f;
#pragma unroll 8
    for (int t = 0; t < TDIM; ++t) acc += ep[t] * sp[t];
    chart[cellOff(n, n + 1, b) + a] = (__bf16)acc;
}

// ------------------------------------------------- staging helper
// Coalesced u16 gathers from chart + vector b128 LDS stores.
// Task = (row, 8-wide k chunk); consecutive lanes take consecutive rows ->
// consecutive 2B elements of the same chart cell (coalesced).
template <int KC>   // k chunks of 8: 4 (Kpad=32) or 8 (Kpad=64)
__device__ __forceinline__ void stageLR(const __bf16* __restrict__ chart,
                                        __bf16* __restrict__ LtB,
                                        __bf16* __restrict__ RtB,
                                        int m, int s, int splits, int h, int tid) {
    const int cbase = m * (NSEQ + 1) * BATCH * KDIM;  // unused; kept simple below
    (void)cbase;
    const int ltTasks = 32 * KC;
    const int total   = ltTasks + 64 * KC;            // 384 or 768: static
#pragma unroll
    for (int t0 = 0; t0 < total; t0 += 128) {
        const int t = t0 + tid;
        Pack pk;
        if (t < ltTasks) {                            // Lt[x][k] = L[k][x]
            const int x  = t & 31;
            const int kc = t >> 5;
            const int xe = 32 * h + x;
#pragma unroll
            for (int j = 0; j < 8; ++j) {
                const int k = kc * 8 + j;
                pk.h[j] = (k < splits)
                    ? chart[cellOff(m, m + 1 + k, 0) + xe]   // b folded by caller
                    : (__bf16)0.f;
            }
            *(uint4*)&LtB[x * LT_S + kc * 8] = pk.q;
        } else {                                      // Rt[y][k] = R[k][y]
            const int t2 = t - ltTasks;
            const int y  = t2 & 63;
            const int kc = t2 >> 6;
#pragma unroll
            for (int j = 0; j < 8; ++j) {
                const int k = kc * 8 + j;
                pk.h[j] = (k < splits)
                    ? chart[cellOff(m + 1 + k, m + s, 0) + y]
                    : (__bf16)0.f;
            }
            *(uint4*)&RtB[y * RT_S + kc * 8] = pk.q;
        }
    }
}

// ------------------------------------------------- one CKY level
// Workgroup = 2 starts (m0, m0+1): 16 rows (2 cells x 8 batch).
// Phase A: pair[r][x][y] = sum_k Lt[x][k] * Rt[y][k]   (WMMA, inner = splits)
// Phase B: vals[r][a]    = sum_xy pair[r][xy] * rule_tf[a][xy] (WMMA, inner 4096)
template <int KSTEPS>   // 1 (s<=33) or 2 (s>=34): static k-loop trip counts
__global__ __launch_bounds__(128)
void level_kernel(__bf16* __restrict__ chart,
                  const __bf16* __restrict__ rule_tf,
                  int s, int mcount) {
    __shared__ __bf16 Lt2[2][32 * LT_S];      // 2 x 4.5 KB (ping-pong)
    __shared__ __bf16 Rt2[2][64 * RT_S];      // 2 x 9.0 KB (ping-pong)
    __shared__ __bf16 pairLDS[16 * PR_S];     // 64.3 KB (inner order y*32+x)

    const int tid  = threadIdx.x;
    const int lane = tid & 31;
    const int wave = tid >> 5;
    const int m0   = blockIdx.x * 2;
    const int splits = s - 1;
    const int hi8  = (lane >> 4) << 3;        // +8 for upper half-lanes
    const int hi16 = (lane >> 4) << 4;        // +16 for upper half-lanes

    f32x8 acc = {};   // this wave's vals tile: rows r=0..15, cols a=16*wave..+15
    const int acol = 16 * wave + (lane & 15);

    for (int h = 0; h < 2; ++h) {             // x in [32h, 32h+32)
        // prologue: stage cb=0 into buffer 0
        {
            const int b0 = 0, m = m0;
            if (m < mcount)
                stageLR<KSTEPS * 4>(chart + b0 * KDIM, Lt2[0], Rt2[0],
                                    m, s, splits, h, tid);
        }
        __syncthreads();
        for (int cb = 0; cb < 16; ++cb) {     // cell c = cb>>3, batch b = cb&7
            const int cur = cb & 1;
            const int c = cb >> 3, b = cb & 7;
            const int m = m0 + c;
            const bool active = (m < mcount);
            // stage NEXT (cb+1) into the other buffer, overlapped with compute
            if (cb + 1 < 16) {
                const int cn = (cb + 1) >> 3, bn = (cb + 1) & 7;
                const int mn = m0 + cn;
                if (mn < mcount)
                    stageLR<KSTEPS * 4>(chart + bn * KDIM, Lt2[cur ^ 1], Rt2[cur ^ 1],
                                        mn, s, splits, h, tid);
            }
            // -------- phase A: 8 tiles this half (xt in {0,1}, yt in 0..3)
            const __bf16* LtB = Lt2[cur];
            const __bf16* RtB = Rt2[cur];
#pragma unroll
            for (int tIt = 0; tIt < 2; ++tIt) {
                const int tId = wave + 4 * tIt;
                const int xt = tId >> 2, yt = tId & 3;
                f32x8 pacc = {};
                if (active) {
#pragma unroll
                    for (int ks = 0; ks < KSTEPS; ++ks) {
                        const int kb = ks * 32;
                        Frag fa, fb;
                        const __bf16* ap =
                            &LtB[(16 * xt + (lane & 15)) * LT_S + kb + hi8];
                        fa.u[0] = *(const uint4*)ap;
                        fa.u[1] = *(const uint4*)(ap + 16);
                        const __bf16* bp =
                            &RtB[(16 * yt + (lane & 15)) * RT_S + kb + hi8];
                        fb.u[0] = *(const uint4*)bp;
                        fb.u[1] = *(const uint4*)(bp + 16);
                        pacc = __builtin_amdgcn_wmma_f32_16x16x32_bf16(
                            false, fa.v, false, fb.v, (short)0, pacc, false, false);
                    }
                }
                // D tile -> pairLDS, inner order c = y*32 + x_local:
                // lane holds 8 consecutive x for one y -> ONE packed b128 store.
                Pack pk;
#pragma unroll
                for (int v = 0; v < 8; ++v) pk.h[v] = (__bf16)pacc[v];
                const int y = 16 * yt + (lane & 15);
                *(uint4*)&pairLDS[cb * PR_S + y * 32 + 16 * xt + hi8] = pk.q;
            }
            // single barrier: publishes stage(cb+1), protects buffer reuse
            __syncthreads();
        }
        // -------- phase B partial over this xy half: inner 2048 -> 64 k-steps
        for (int ks = 0; ks < 64; ++ks) {
            const int kb = ks * 32;
            Frag fa, fb;
            const __bf16* ap = &pairLDS[(lane & 15) * PR_S + kb + hi8];
            fa.u[0] = *(const uint4*)ap;
            fa.u[1] = *(const uint4*)(ap + 16);
            const __bf16* bp = rule_tf + acol * 4096 + h * 2048 + kb + hi8;
            fb.u[0] = *(const uint4*)bp;
            fb.u[1] = *(const uint4*)(bp + 16);
            acc = __builtin_amdgcn_wmma_f32_16x16x32_bf16(
                false, fa.v, false, fb.v, (short)0, acc, false, false);
        }
        __syncthreads();   // pairLDS reuse safety for next h
    }
    // -------- store vals: (r, a), r = c*8 + b
#pragma unroll
    for (int v = 0; v < 8; ++v) {
        const int r = v + hi8;
        const int c = r >> 3, b = r & 7;
        const int m = m0 + c;
        if (m < mcount)
            chart[cellOff(m, m + s, b) + acol] = (__bf16)acc[v];
    }
}

// ------------------------------------------------- final dot
__global__ void final_kernel(const __bf16* __restrict__ chart,
                             const float* __restrict__ root,
                             float* __restrict__ out) {
    __shared__ float red[64];
    const int tid = threadIdx.x;   // 64 threads = nonterminal a
    for (int b = 0; b < BATCH; ++b) {
        red[tid] = root[tid] * (float)chart[cellOff(0, NSEQ, b) + tid];
        __syncthreads();
        for (int st = 32; st > 0; st >>= 1) {
            if (tid < st) red[tid] += red[tid + st];
            __syncthreads();
        }
        if (tid == 0) out[b] = red[0];
        __syncthreads();
    }
}

extern "C" void kernel_launch(void* const* d_in, const int* in_sizes, int n_in,
                              void* d_out, int out_size, void* d_ws, size_t ws_size,
                              hipStream_t stream) {
    const float* seq  = (const float*)d_in[0];   // (8,64,128)
    const float* rule = (const float*)d_in[1];   // (64,64,64)
    const float* emit = (const float*)d_in[2];   // (64,128)
    const float* root = (const float*)d_in[3];   // (64,)
    float* out = (float*)d_out;                  // (8,)

    __bf16* rule_tf = (__bf16*)d_ws;                           // 512 KB
    __bf16* chart   = (__bf16*)((char*)d_ws + 512 * 1024);     // 65*65*8*64 bf16

    convert_rule_kernel<<<dim3(1024), dim3(256), 0, stream>>>(rule, rule_tf);
    span1_kernel<<<dim3(NSEQ, BATCH), dim3(KDIM), 0, stream>>>(seq, emit, chart);
    for (int s = 2; s <= NSEQ; ++s) {
        const int mcount = NSEQ - s + 1;
        const int blocks = (mcount + 1) / 2;
        if (s <= 33)   // splits <= 32 -> Kpad = 32 -> 1 k-step in phase A
            level_kernel<1><<<dim3(blocks), dim3(128), 0, stream>>>(chart, rule_tf, s, mcount);
        else           // Kpad = 64 -> 2 k-steps
            level_kernel<2><<<dim3(blocks), dim3(128), 0, stream>>>(chart, rule_tf, s, mcount);
    }
    final_kernel<<<dim3(1), dim3(64), 0, stream>>>(chart, root, out);
}
